// OGBlayer_79276506349851
// MI455X (gfx1250) — matrix operand order repaired and checked by reading, not compiled, not run on hardware
//
#include <hip/hip_runtime.h>
#include <hip/hip_bf16.h>

// ---------------------------------------------------------------------------
// Fused GNN layer for MI455X (gfx1250, wave32, WMMA).
//   k1: h = node_feats @ Wn^T + bn  -> ws ;  d_out = relu(h + res_w)/degs
//   k2: e = edge_feats @ We^T + be  ;  atomicAdd(d_out[dst], norm*relu(h[src]+e))
// GEMM: bf16 hi/lo split (3 WMMA terms) for ~fp32 accuracy, f32 accumulate.
// ---------------------------------------------------------------------------

typedef __bf16 v16bf __attribute__((ext_vector_type(16)));
typedef __bf16 v8bf  __attribute__((ext_vector_type(8)));
typedef float  v8f   __attribute__((ext_vector_type(8)));
typedef float  v4f   __attribute__((ext_vector_type(4)));

#define D128       128
#define LDS_STRIDE 136   // padded bf16 row stride: 68 dwords -> bank rotate by 4/row

// Stage 128x128 fp32 weight matrix into LDS as bf16 hi + bf16 lo (split).
__device__ __forceinline__ void stage_weights_split(const float* __restrict__ W,
                                                    __bf16* sHi, __bf16* sLo, int tid) {
  for (int idx = tid; idx < D128 * D128; idx += 256) {
    int r = idx >> 7, c = idx & 127;
    float w  = W[idx];
    __bf16 h = (__bf16)w;
    sHi[r * LDS_STRIDE + c] = h;
    sLo[r * LDS_STRIDE + c] = (__bf16)(w - (float)h);
  }
}

// One wave computes a 16-row strip of X (row-major, stride 128) times W^T
// (W staged in LDS as N x K bf16 hi/lo), producing acc[8] = 16x128 f32.
template <bool NT>
__device__ __forceinline__ void gemm_strip(const float* __restrict__ X, long row0,
                                           const __bf16* sHi, const __bf16* sLo,
                                           int lane, v8f acc[8]) {
  const int m  = lane & 15;
  const int ka = (lane >> 4) << 3;   // A-matrix per-lane K base: 0 or 8
  const int kb = (lane >> 4) << 3;   // B-matrix per-lane K base: 0 or 8
  const float* xr = X + (size_t)(row0 + m) * D128;

#pragma unroll
  for (int ks = 0; ks < 4; ++ks) {   // K = 128 in steps of 32
    const float* p = xr + ks * 32 + ka;
    v4f a0, a1, a2, a3;
    if (NT) {
      a0 = __builtin_nontemporal_load((const v4f*)(p));
      a1 = __builtin_nontemporal_load((const v4f*)(p + 4));
      a2 = __builtin_nontemporal_load((const v4f*)(p + 16));
      a3 = __builtin_nontemporal_load((const v4f*)(p + 20));
    } else {
      a0 = *(const v4f*)(p);
      a1 = *(const v4f*)(p + 4);
      a2 = *(const v4f*)(p + 16);
      a3 = *(const v4f*)(p + 20);
    }
    float av[16] = {a0[0], a0[1], a0[2], a0[3], a1[0], a1[1], a1[2], a1[3],
                    a2[0], a2[1], a2[2], a2[3], a3[0], a3[1], a3[2], a3[3]};
    v16bf ahi, alo;
#pragma unroll
    for (int i = 0; i < 16; ++i) {
      __bf16 h = (__bf16)av[i];
      ahi[i] = h;
      alo[i] = (__bf16)(av[i] - (float)h);
    }

#pragma unroll
    for (int t = 0; t < 8; ++t) {    // 8 N-tiles of 16
      const __bf16* bph = sHi + (t * 16 + m) * LDS_STRIDE + ks * 32 + kb;
      const __bf16* bpl = sLo + (t * 16 + m) * LDS_STRIDE + ks * 32 + kb;
      v8bf bh0 = *(const v8bf*)(bph);
      v8bf bh1 = *(const v8bf*)(bph + 16);
      v8bf bl0 = *(const v8bf*)(bpl);
      v8bf bl1 = *(const v8bf*)(bpl + 16);
      v16bf bhi, blo;
#pragma unroll
      for (int i = 0; i < 8; ++i) {
        bhi[i] = bh0[i]; bhi[8 + i] = bh1[i];
        blo[i] = bl0[i]; blo[8 + i] = bl1[i];
      }
      acc[t] = __builtin_amdgcn_wmma_f32_16x16x32_bf16(false, ahi, false, bhi,
                                                       (short)0, acc[t], false, false);
      acc[t] = __builtin_amdgcn_wmma_f32_16x16x32_bf16(false, alo, false, bhi,
                                                       (short)0, acc[t], false, false);
      acc[t] = __builtin_amdgcn_wmma_f32_16x16x32_bf16(false, ahi, false, blo,
                                                       (short)0, acc[t], false, false);
    }
  }
}

// ---------------------------------------------------------------------------
// Kernel 1: node projection + residual. Writes h to ws and initializes d_out.
// ---------------------------------------------------------------------------
__global__ void __launch_bounds__(256)
ogb_node_kernel(const float* __restrict__ node_feats, const float* __restrict__ Wn,
                const float* __restrict__ bn, const float* __restrict__ res_w,
                const float* __restrict__ degs, float* __restrict__ h_ws,
                float* __restrict__ out, int nstrips) {
  __shared__ __align__(16) __bf16 sHi[D128 * LDS_STRIDE];
  __shared__ __align__(16) __bf16 sLo[D128 * LDS_STRIDE];
  const int tid = threadIdx.x;
  stage_weights_split(Wn, sHi, sLo, tid);
  __syncthreads();

  const int wave = tid >> 5, lane = tid & 31;
  const int strip = blockIdx.x * 8 + wave;
  if (strip >= nstrips) return;          // wave-uniform: EXEC stays all-ones
  const long row0 = (long)strip * 16;

  v8f acc[8] = {};
  gemm_strip<false>(node_feats, row0, sHi, sLo, lane, acc);

  const int m = lane & 15;
  const int mbase = (lane < 16) ? 0 : 8;
#pragma unroll
  for (int t = 0; t < 8; ++t) {
    const int n = t * 16 + m;
    const float bias = bn[n];
    const float rw   = res_w[n];
#pragma unroll
    for (int v = 0; v < 8; ++v) {
      const long r = row0 + mbase + v;
      const float hv = acc[t][v] + bias;
      h_ws[(size_t)r * D128 + n] = hv;
      out[(size_t)r * D128 + n]  = fmaxf(hv + rw, 0.0f) / degs[r];
    }
  }
}

// ---------------------------------------------------------------------------
// Kernel 2: edge projection + gather h[src] + fused relu/scale + scatter-add.
// ---------------------------------------------------------------------------
__global__ void __launch_bounds__(256)
ogb_edge_kernel(const float* __restrict__ edge_feats, const float* __restrict__ We,
                const float* __restrict__ be, const float* __restrict__ norm,
                const int* __restrict__ src, const int* __restrict__ dst,
                const float* __restrict__ h_ws, float* __restrict__ out, int estrips) {
  __shared__ __align__(16) __bf16 sHi[D128 * LDS_STRIDE];
  __shared__ __align__(16) __bf16 sLo[D128 * LDS_STRIDE];
  const int tid = threadIdx.x;
  stage_weights_split(We, sHi, sLo, tid);
  __syncthreads();

  const int wave = tid >> 5, lane = tid & 31;
  const int strip = blockIdx.x * 8 + wave;
  if (strip >= estrips) return;
  const long row0 = (long)strip * 16;

  v8f acc[8] = {};
  gemm_strip<true>(edge_feats, row0, sHi, sLo, lane, acc);  // NT: stream, keep h in L2

  const int m = lane & 15;
  const int mbase = (lane < 16) ? 0 : 8;

  int   sv[8]; int dv[8]; float nv[8];
#pragma unroll
  for (int v = 0; v < 8; ++v) {
    const long e = row0 + mbase + v;
    sv[v] = src[e];
    dv[v] = dst[e];
    nv[v] = norm[e];
  }

#pragma unroll
  for (int t = 0; t < 8; ++t) {
    const int n = t * 16 + m;
    const float bias = be[n];
#pragma unroll
    for (int v = 0; v < 8; ++v) {
      const float ev  = acc[t][v] + bias;                       // e = X_e·We^T + be
      const float hv  = h_ws[(size_t)sv[v] * D128 + n];         // h[src] gather (L2)
      const float msg = nv[v] * fmaxf(hv + ev, 0.0f);           // norm * relu(h[src]+e)
      atomicAdd(out + (size_t)dv[v] * D128 + n, msg);           // segment-sum scatter
    }
  }
}

// ---------------------------------------------------------------------------
extern "C" void kernel_launch(void* const* d_in, const int* in_sizes, int n_in,
                              void* d_out, int out_size, void* d_ws, size_t ws_size,
                              hipStream_t stream) {
  const float* node_feats = (const float*)d_in[0];
  const float* edge_feats = (const float*)d_in[1];
  const float* degs       = (const float*)d_in[2];
  const float* norm       = (const float*)d_in[3];
  const int*   src        = (const int*)d_in[4];
  const int*   dst        = (const int*)d_in[5];
  const float* Wn         = (const float*)d_in[6];
  const float* bn         = (const float*)d_in[7];
  const float* We         = (const float*)d_in[8];
  const float* be         = (const float*)d_in[9];
  const float* res_w      = (const float*)d_in[10];

  const int N = in_sizes[0] / D128;      // 20000 (multiple of 16)
  const int E = in_sizes[1] / D128;      // 640000 (multiple of 16)

  float* h_ws = (float*)d_ws;            // N*128*4 = 10.24 MB scratch
  float* out  = (float*)d_out;

  const int nstrips = (N + 15) / 16;
  const int nblocks = (nstrips + 7) / 8;
  ogb_node_kernel<<<nblocks, 256, 0, stream>>>(node_feats, Wn, bn, res_w, degs,
                                               h_ws, out, nstrips);

  const int estrips = E / 16;
  const int eblocks = (estrips + 7) / 8;
  ogb_edge_kernel<<<eblocks, 256, 0, stream>>>(edge_feats, We, be, norm, src, dst,
                                               h_ws, out, estrips);
}
